// STGAT_39539468927348
// MI455X (gfx1250) — matrix operand order repaired and checked by reading, not compiled, not run on hardware
//
#include <hip/hip_runtime.h>
#include <hip/hip_bf16.h>
#include <math.h>

// ---------- problem constants (from reference) ----------
#define S_GRAPHS 1024
#define NPG      64
#define N_NODES  65536       // S_GRAPHS * NPG
#define E_TOT    524288
#define EPG      512         // edges per graph
#define EFF      576         // EPG + NPG self loops
#define NF       32
#define EF       16
#define HC       64
#define H1       3
#define F1       192         // H1*HC

typedef __attribute__((ext_vector_type(16))) _Float16 v16h;
typedef __attribute__((ext_vector_type(8)))  float    v8f;
typedef __attribute__((ext_vector_type(4)))  unsigned int u32x4;
typedef __attribute__((ext_vector_type(8)))  int      i32x8;
typedef __attribute__((ext_vector_type(4)))  int      i32x4;

// =============== Tensor Data Mover: 1-D contiguous global->LDS copy =================
// D# built per CDNA5 ISA ch.8: group0 = {flags, lds_addr, global_addr lo/hi|type=2},
// group1 = {data_size=4B, tensor_dim0=n, tensor_dim1=1, tile_dim0=n, stride=n}.
__device__ inline void tdm_load_1d(const void* gsrc, void* lds_dst, unsigned n /* 4B elems, <=65535 */) {
  unsigned long long ga = (unsigned long long)(size_t)gsrc;
  unsigned la = (unsigned)(size_t)lds_dst;          // LDS aperture: offset in addr[31:0]
  u32x4 g0;
  g0[0] = 1u;                                        // count=1, user D#, no gather
  g0[1] = la;                                        // lds_addr
  g0[2] = (unsigned)(ga & 0xFFFFFFFFu);              // global_addr[31:0]
  g0[3] = (unsigned)((ga >> 32) & 0x1FFFFFFu) | (2u << 30);  // global_addr[56:32] | type=2
  i32x8 g1;
  g1[0] = (int)(2u << 16);                           // data_size = 4 bytes
  g1[1] = (int)((n & 0xFFFFu) << 16);                // tensor_dim0[15:0]
  g1[2] = (int)((n >> 16) & 0xFFFFu) | (1 << 16);    // tensor_dim0[31:16] | tensor_dim1=1 lo
  g1[3] = (int)((n & 0xFFFFu) << 16);                // tensor_dim1 hi(0) | tile_dim0 = n
  g1[4] = 0;                                         // tile_dim1 = 0 (1-D), tile_dim2 = 0
  g1[5] = (int)n;                                    // tensor_dim0_stride[31:0]
  g1[6] = 0;                                         // stride hi | dim1_stride lo
  g1[7] = 0;
  i32x4 z4 = {0, 0, 0, 0};
#if __clang_major__ >= 23
  i32x8 z8 = {0, 0, 0, 0, 0, 0, 0, 0};
  __builtin_amdgcn_tensor_load_to_lds(g0, g1, z4, z4, z8, 0);
#else
  __builtin_amdgcn_tensor_load_to_lds(g0, g1, z4, z4, 0);
#endif
}

// =============== WMMA fragment helpers (gfx1250 wave32 layouts) =====================
// A fragment 16x32 f16: lane l<16: M=l, elems 0..7=K0..7, 8..15=K16..23
//                       lane l>=16: M=l-16, elems 0..7=K8..15, 8..15=K24..31
__device__ inline v16h load_a_frag_v(const float* __restrict__ A, int lda, int row0, int k0, int lane) {
  const int r   = lane & 15;
  const int kb0 = (lane < 16) ? 0 : 8;
  const int kb1 = (lane < 16) ? 16 : 24;
  const float* base = A + (size_t)(row0 + r) * lda + k0;
  const float4* p0 = (const float4*)(base + kb0);
  const float4* p1 = (const float4*)(base + kb1);
  float4 a0 = p0[0], a1 = p0[1], b0 = p1[0], b1 = p1[1];
  v16h a;
  a[0]  = (_Float16)a0.x; a[1]  = (_Float16)a0.y; a[2]  = (_Float16)a0.z; a[3]  = (_Float16)a0.w;
  a[4]  = (_Float16)a1.x; a[5]  = (_Float16)a1.y; a[6]  = (_Float16)a1.z; a[7]  = (_Float16)a1.w;
  a[8]  = (_Float16)b0.x; a[9]  = (_Float16)b0.y; a[10] = (_Float16)b0.z; a[11] = (_Float16)b0.w;
  a[12] = (_Float16)b1.x; a[13] = (_Float16)b1.y; a[14] = (_Float16)b1.z; a[15] = (_Float16)b1.w;
  return a;
}

// B fragment 32x16 f16 (K x N) gathered from a row-major f32 matrix (used only by pack kernel)
__device__ inline v16h gather_b_frag(const float* __restrict__ B, int ldb, int k0, int col0, int lane) {
  v16h b;
  const int c  = lane & 15;
  const int kb = (lane < 16) ? 0 : 8;
  const float* p = B + (size_t)(k0 + kb) * ldb + col0 + c;
#pragma unroll
  for (int i = 0; i < 16; ++i) b[i] = (_Float16)p[(size_t)(i < 8 ? i : i + 8) * ldb];
  return b;
}

// C/D fragment: VGPR r -> lanes 0..15: (M=r, N=lane); lanes 16..31: (M=r+8, N=lane-16)
__device__ inline void store_c_bias(float* __restrict__ C, int ldc, int row0, int col0,
                                    v8f c, int lane, const float* __restrict__ bias) {
  const int col   = lane & 15;
  const int rbase = (lane < 16) ? 0 : 8;
  const float bv  = bias[col0 + col];
#pragma unroll
  for (int r = 0; r < 8; ++r)
    C[(size_t)(row0 + rbase + r) * ldc + col0 + col] = c[r] + bv;
}

// =============== weight pack: f32 -> WMMA-ready f16 fragments =======================
// frag ids: [0,12)=Wl1 tiles, [12,24)=Wr1 tiles, [24,48)=Wl2 (j*6+kk), [48,72)=Wr2
__global__ void __launch_bounds__(32)
stgat_pack_w(const float* __restrict__ Wl1, const float* __restrict__ Wr1,
             const float* __restrict__ Wl2, const float* __restrict__ Wr2,
             _Float16* __restrict__ out) {
  const int f = blockIdx.x, lane = threadIdx.x;
  const float* W; int ld, k0, col0;
  if (f < 24) {
    W = (f < 12) ? Wl1 : Wr1; ld = F1; k0 = 0; col0 = (f % 12) * 16;
  } else {
    const int f2  = f - 24;
    const int idx = f2 % 24;
    W = (f2 < 24) ? Wl2 : Wr2; ld = HC; col0 = (idx / 6) * 16; k0 = (idx % 6) * 32;
  }
  v16h b = gather_b_frag(W, ld, k0, col0, lane);
  *(v16h*)(out + ((size_t)f * 32 + lane) * 16) = b;
}

__device__ inline v16h load_packed_frag(const _Float16* __restrict__ wp, int frag, int lane) {
  return *(const v16h*)(wp + ((size_t)frag * 32 + lane) * 16);
}

// =============== GEMM 1: xl1 = x@Wl1+bl1, xr1 = x@Wr1+br1 (K=32, 1 WMMA/tile) =======
__global__ void __launch_bounds__(128)
stgat_gemm1(const float* __restrict__ x, const _Float16* __restrict__ wp,
            const float* __restrict__ bl, const float* __restrict__ br,
            float* __restrict__ xl, float* __restrict__ xr) {
  const int lane = threadIdx.x & 31;
  const int wave = threadIdx.x >> 5;
  const int row0 = (blockIdx.x * 4 + wave) * 16;
  v16h a = load_a_frag_v(x, NF, row0, 0, lane);
#pragma unroll
  for (int j = 0; j < F1 / 16; ++j) {
    v16h b = load_packed_frag(wp, j, lane);
    v8f  c = {};
    c = __builtin_amdgcn_wmma_f32_16x16x32_f16(false, a, false, b, (short)0, c, false, false);
    store_c_bias(xl, F1, row0, j * 16, c, lane, bl);
  }
#pragma unroll
  for (int j = 0; j < F1 / 16; ++j) {
    v16h b = load_packed_frag(wp, 12 + j, lane);
    v8f  c = {};
    c = __builtin_amdgcn_wmma_f32_16x16x32_f16(false, a, false, b, (short)0, c, false, false);
    store_c_bias(xr, F1, row0, j * 16, c, lane, br);
  }
}

// =============== GEMM 2: xl2 = h1@Wl2+bl2, xr2 = h1@Wr2+br2 (K=192, 6 WMMAs) =======
__global__ void __launch_bounds__(128)
stgat_gemm2(const float* __restrict__ h, const _Float16* __restrict__ wp,
            const float* __restrict__ bl, const float* __restrict__ br,
            float* __restrict__ xl, float* __restrict__ xr) {
  const int lane = threadIdx.x & 31;
  const int wave = threadIdx.x >> 5;
  const int row0 = (blockIdx.x * 4 + wave) * 16;
  v16h a[6];
#pragma unroll
  for (int kk = 0; kk < 6; ++kk) a[kk] = load_a_frag_v(h, F1, row0, kk * 32, lane);
#pragma unroll
  for (int j = 0; j < HC / 16; ++j) {
    v8f c = {};
#pragma unroll
    for (int kk = 0; kk < 6; ++kk) {
      v16h b = load_packed_frag(wp, 24 + j * 6 + kk, lane);
      c = __builtin_amdgcn_wmma_f32_16x16x32_f16(false, a[kk], false, b, (short)0, c, false, false);
    }
    store_c_bias(xl, HC, row0, j * 16, c, lane, bl);
  }
#pragma unroll
  for (int j = 0; j < HC / 16; ++j) {
    v8f c = {};
#pragma unroll
    for (int kk = 0; kk < 6; ++kk) {
      v16h b = load_packed_frag(wp, 48 + j * 6 + kk, lane);
      c = __builtin_amdgcn_wmma_f32_16x16x32_f16(false, a[kk], false, b, (short)0, c, false, false);
    }
    store_c_bias(xr, HC, row0, j * 16, c, lane, br);
  }
}

// =============== GATv2 attention layer 1 (H=3, C=64), one workgroup per graph =======
__global__ void __launch_bounds__(256)
stgat_attn1(const float* __restrict__ xl1, const float* __restrict__ xr1,
            const int* __restrict__ EI, const float* __restrict__ EA,
            const float* __restrict__ We, const float* __restrict__ att,
            const float* __restrict__ bias, float* __restrict__ h1out) {
  const int g = blockIdx.x, tid = threadIdx.x, bd = blockDim.x;
  const int base = g * NPG;

  extern __shared__ float smem[];
  float* sXl    = smem;                    // 64*192
  float* sXr    = sXl + NPG * F1;          // 64*192
  float* sWeT   = sXr + NPG * F1;          // 192*16 (transposed: [f][k])
  float* sAtt   = sWeT + F1 * EF;          // 192
  float* sLoop  = sAtt + F1;               // 64*16
  float* sDeg   = sLoop + NPG * EF;        // 64
  float* sScore = sDeg + NPG;              // 3*576 (scores -> exp -> alpha)
  float* sSum   = sScore + H1 * EFF;       // 3*64
  int*   sEdge  = (int*)(sSum + H1 * NPG); // 576
  int*   sList  = sEdge + EFF;             // 576 CSR edge ids
  int*   sCnt   = sList + EFF;             // 64
  int*   sOff   = sCnt + NPG;              // 65
  int*   sCur   = sOff + NPG + 1;          // 64

  const float* xlg = xl1 + (size_t)base * F1;
  const float* xrg = xr1 + (size_t)base * F1;

  // async TDM staging of xl/xr tiles while we build the edge tables
  if (tid < 32) {
    tdm_load_1d(xlg, sXl, NPG * F1);
    tdm_load_1d(xrg, sXr, NPG * F1);
  }

  for (int i = tid; i < F1 * EF; i += bd) sWeT[i] = We[(i & 15) * F1 + (i >> 4)];
  for (int i = tid; i < F1; i += bd) sAtt[i] = att[i];
  for (int i = tid; i < NPG * EF; i += bd) sLoop[i] = 0.f;
  for (int i = tid; i < NPG; i += bd) { sDeg[i] = 0.f; sCnt[i] = 0; }
  __syncthreads();

  // edge table, CSR counts, degree, self-loop attr sums (fill_value='mean')
  for (int e = tid; e < EFF; e += bd) {
    int s, d;
    if (e < EPG) {
      const size_t ge = (size_t)g * EPG + e;
      s = EI[ge] - base;
      d = EI[(size_t)E_TOT + ge] - base;
      atomicAdd(&sDeg[d], 1.f);
      const float4* p4 = (const float4*)(EA + ge * EF);
      float4 q0 = p4[0], q1 = p4[1], q2 = p4[2], q3 = p4[3];
      float qa[EF] = {q0.x, q0.y, q0.z, q0.w, q1.x, q1.y, q1.z, q1.w,
                      q2.x, q2.y, q2.z, q2.w, q3.x, q3.y, q3.z, q3.w};
#pragma unroll
      for (int k = 0; k < EF; ++k) atomicAdd(&sLoop[d * EF + k], qa[k]);
    } else {
      s = d = e - EPG;
    }
    sEdge[e] = s | (d << 8);
    atomicAdd(&sCnt[d], 1);
  }
  __syncthreads();
  for (int i = tid; i < NPG * EF; i += bd) sLoop[i] /= fmaxf(sDeg[i >> 4], 1.f);
  if (tid == 0) {           // CSR prefix
    int run = 0;
    for (int n = 0; n < NPG; ++n) { sOff[n] = run; sCur[n] = run; run += sCnt[n]; }
    sOff[NPG] = run;
  }
  __syncthreads();
  for (int e = tid; e < EFF; e += bd) {
    const int d = (sEdge[e] >> 8) & 255;
    sList[atomicAdd(&sCur[d], 1)] = e;
  }
  // make sure TDM data has landed before the score pass
  if (tid < 32) __builtin_amdgcn_s_wait_tensorcnt((short)0);
  __syncthreads();

  // scores: score[e,h] = sum_c lrelu(xl[s]+xr[d]+ea@We)_hc * att[h,c]
  for (int eh = tid; eh < H1 * EFF; eh += bd) {
    const int e  = eh % EFF;
    const int hh = eh / EFF;
    const int s  = sEdge[e] & 255, d = (sEdge[e] >> 8) & 255;
    float ea[EF];
    if (e < EPG) {
      const float4* p4 = (const float4*)(EA + ((size_t)g * EPG + e) * EF);
      float4 q0 = p4[0], q1 = p4[1], q2 = p4[2], q3 = p4[3];
      ea[0] = q0.x; ea[1] = q0.y; ea[2]  = q0.z; ea[3]  = q0.w;
      ea[4] = q1.x; ea[5] = q1.y; ea[6]  = q1.z; ea[7]  = q1.w;
      ea[8] = q2.x; ea[9] = q2.y; ea[10] = q2.z; ea[11] = q2.w;
      ea[12] = q3.x; ea[13] = q3.y; ea[14] = q3.z; ea[15] = q3.w;
    } else {
#pragma unroll
      for (int k = 0; k < EF; ++k) ea[k] = sLoop[s * EF + k];
    }
    const float* xlr = sXl + s * F1 + hh * HC;
    const float* xrr = sXr + d * F1 + hh * HC;
    const float* atr = sAtt + hh * HC;
    float sc = 0.f;
    for (int c = 0; c < HC; ++c) {
      const float* w = sWeT + (hh * HC + c) * EF;
      float ep = 0.f;
#pragma unroll
      for (int k = 0; k < EF; ++k) ep += ea[k] * w[k];
      float m = xlr[c] + xrr[c] + ep;
      m = (m > 0.f) ? m : 0.2f * m;
      sc += m * atr[c];
    }
    sScore[hh * EFF + e] = sc;
  }
  __syncthreads();

  // segment softmax via CSR: owner thread per (head, dst); store alpha in sScore
  for (int p = tid; p < H1 * NPG; p += bd) {
    const int hh = p / NPG, n = p % NPG;
    const int b0 = sOff[n], b1 = sOff[n + 1];
    float mx = -1e30f;
    for (int i = b0; i < b1; ++i) mx = fmaxf(mx, sScore[hh * EFF + sList[i]]);
    float sm = 0.f;
    for (int i = b0; i < b1; ++i) {
      const float v = __expf(sScore[hh * EFF + sList[i]] - mx);
      sScore[hh * EFF + sList[i]] = v;
      sm += v;
    }
    const float inv = 1.f / (sm + 1e-16f);
    for (int i = b0; i < b1; ++i) sScore[hh * EFF + sList[i]] *= inv;
  }
  __syncthreads();

  // owner-computes aggregation (no atomics), fused bias+ELU, straight to global
  float* og = h1out + (size_t)base * F1;
  for (int w = tid; w < NPG * H1 * 4; w += bd) {
    const int n  = w / (H1 * 4);
    const int r  = w - n * (H1 * 4);
    const int hh = r >> 2;
    const int q  = (r & 3) << 4;
    float acc[16];
#pragma unroll
    for (int t = 0; t < 16; ++t) acc[t] = 0.f;
    for (int i = sOff[n]; i < sOff[n + 1]; ++i) {
      const int e = sList[i];
      const int s = sEdge[e] & 255;
      const float a = sScore[hh * EFF + e];
      const float* xp = sXl + s * F1 + hh * HC + q;
#pragma unroll
      for (int t = 0; t < 16; ++t) acc[t] += a * xp[t];
    }
    float* op = og + n * F1 + hh * HC + q;
    const float* bp = bias + hh * HC + q;
#pragma unroll
    for (int t = 0; t < 16; ++t) {
      const float v = acc[t] + bp[t];
      op[t] = (v > 0.f) ? v : (__expf(v) - 1.f);
    }
  }
}

// =============== GATv2 attention layer 2 (H=1, C=64) + mean pooling =================
__global__ void __launch_bounds__(256)
stgat_attn2(const float* __restrict__ xl2, const float* __restrict__ xr2,
            const int* __restrict__ EI, const float* __restrict__ EA,
            const float* __restrict__ We, const float* __restrict__ att,
            const float* __restrict__ bias, float* __restrict__ pooled) {
  const int g = blockIdx.x, tid = threadIdx.x, bd = blockDim.x;
  const int base = g * NPG;

  extern __shared__ float smem[];
  float* sXl    = smem;                    // 64*64
  float* sXr    = sXl + NPG * HC;          // 64*64
  float* sWeT   = sXr + NPG * HC;          // 64*16 (transposed)
  float* sAtt   = sWeT + HC * EF;          // 64
  float* sLoop  = sAtt + HC;               // 64*16
  float* sDeg   = sLoop + NPG * EF;        // 64
  float* sScore = sDeg + NPG;              // 576
  float* sSum   = sScore + EFF;            // 64
  float* sPool  = sSum + NPG;              // 64
  int*   sEdge  = (int*)(sPool + HC);      // 576
  int*   sList  = sEdge + EFF;             // 576
  int*   sCnt   = sList + EFF;             // 64
  int*   sOff   = sCnt + NPG;              // 65
  int*   sCur   = sOff + NPG + 1;          // 64

  const float* xlg = xl2 + (size_t)base * HC;
  const float* xrg = xr2 + (size_t)base * HC;

  if (tid < 32) {
    tdm_load_1d(xlg, sXl, NPG * HC);
    tdm_load_1d(xrg, sXr, NPG * HC);
  }

  for (int i = tid; i < HC * EF; i += bd) sWeT[i] = We[(i & 15) * HC + (i >> 4)];
  for (int i = tid; i < HC; i += bd) { sAtt[i] = att[i]; sPool[i] = 0.f; }
  for (int i = tid; i < NPG * EF; i += bd) sLoop[i] = 0.f;
  for (int i = tid; i < NPG; i += bd) { sDeg[i] = 0.f; sCnt[i] = 0; }
  __syncthreads();

  for (int e = tid; e < EFF; e += bd) {
    int s, d;
    if (e < EPG) {
      const size_t ge = (size_t)g * EPG + e;
      s = EI[ge] - base;
      d = EI[(size_t)E_TOT + ge] - base;
      atomicAdd(&sDeg[d], 1.f);
      const float4* p4 = (const float4*)(EA + ge * EF);
      float4 q0 = p4[0], q1 = p4[1], q2 = p4[2], q3 = p4[3];
      float qa[EF] = {q0.x, q0.y, q0.z, q0.w, q1.x, q1.y, q1.z, q1.w,
                      q2.x, q2.y, q2.z, q2.w, q3.x, q3.y, q3.z, q3.w};
#pragma unroll
      for (int k = 0; k < EF; ++k) atomicAdd(&sLoop[d * EF + k], qa[k]);
    } else {
      s = d = e - EPG;
    }
    sEdge[e] = s | (d << 8);
    atomicAdd(&sCnt[d], 1);
  }
  __syncthreads();
  for (int i = tid; i < NPG * EF; i += bd) sLoop[i] /= fmaxf(sDeg[i >> 4], 1.f);
  if (tid == 0) {
    int run = 0;
    for (int n = 0; n < NPG; ++n) { sOff[n] = run; sCur[n] = run; run += sCnt[n]; }
    sOff[NPG] = run;
  }
  __syncthreads();
  for (int e = tid; e < EFF; e += bd) {
    const int d = (sEdge[e] >> 8) & 255;
    sList[atomicAdd(&sCur[d], 1)] = e;
  }
  if (tid < 32) __builtin_amdgcn_s_wait_tensorcnt((short)0);
  __syncthreads();

  for (int e = tid; e < EFF; e += bd) {
    const int s = sEdge[e] & 255, d = (sEdge[e] >> 8) & 255;
    float ea[EF];
    if (e < EPG) {
      const float4* p4 = (const float4*)(EA + ((size_t)g * EPG + e) * EF);
      float4 q0 = p4[0], q1 = p4[1], q2 = p4[2], q3 = p4[3];
      ea[0] = q0.x; ea[1] = q0.y; ea[2]  = q0.z; ea[3]  = q0.w;
      ea[4] = q1.x; ea[5] = q1.y; ea[6]  = q1.z; ea[7]  = q1.w;
      ea[8] = q2.x; ea[9] = q2.y; ea[10] = q2.z; ea[11] = q2.w;
      ea[12] = q3.x; ea[13] = q3.y; ea[14] = q3.z; ea[15] = q3.w;
    } else {
#pragma unroll
      for (int k = 0; k < EF; ++k) ea[k] = sLoop[s * EF + k];
    }
    const float* xlr = sXl + s * HC;
    const float* xrr = sXr + d * HC;
    float sc = 0.f;
    for (int c = 0; c < HC; ++c) {
      const float* w = sWeT + c * EF;
      float ep = 0.f;
#pragma unroll
      for (int k = 0; k < EF; ++k) ep += ea[k] * w[k];
      float m = xlr[c] + xrr[c] + ep;
      m = (m > 0.f) ? m : 0.2f * m;
      sc += m * sAtt[c];
    }
    sScore[e] = sc;
  }
  __syncthreads();

  for (int n = tid; n < NPG; n += bd) {
    const int b0 = sOff[n], b1 = sOff[n + 1];
    float mx = -1e30f;
    for (int i = b0; i < b1; ++i) mx = fmaxf(mx, sScore[sList[i]]);
    float sm = 0.f;
    for (int i = b0; i < b1; ++i) {
      const float v = __expf(sScore[sList[i]] - mx);
      sScore[sList[i]] = v;
      sm += v;
    }
    const float inv = 1.f / (sm + 1e-16f);
    for (int i = b0; i < b1; ++i) sScore[sList[i]] *= inv;
  }
  __syncthreads();

  // aggregation + bias + ELU, then accumulate per-channel pool sums
  for (int w = tid; w < NPG * 4; w += bd) {
    const int n = w >> 2;
    const int q = (w & 3) << 4;
    float acc[16];
#pragma unroll
    for (int t = 0; t < 16; ++t) acc[t] = 0.f;
    for (int i = sOff[n]; i < sOff[n + 1]; ++i) {
      const int e = sList[i];
      const int s = sEdge[e] & 255;
      const float a = sScore[e];
      const float* xp = sXl + s * HC + q;
#pragma unroll
      for (int t = 0; t < 16; ++t) acc[t] += a * xp[t];
    }
#pragma unroll
    for (int t = 0; t < 16; ++t) {
      float v = acc[t] + bias[q + t];
      v = (v > 0.f) ? v : (__expf(v) - 1.f);
      atomicAdd(&sPool[q + t], v);
    }
  }
  __syncthreads();
  for (int c = tid; c < HC; c += bd)
    pooled[(size_t)g * HC + c] = sPool[c] * (1.f / (float)NPG);
}

// =============== LSTM over 1024 steps + MLP head (single block, sequential) =========
__global__ void __launch_bounds__(256)
stgat_lstm_head(const float* __restrict__ pooled,
                const float* __restrict__ Wih, const float* __restrict__ Whh,
                const float* __restrict__ bih, const float* __restrict__ bhh,
                const float* __restrict__ W1, const float* __restrict__ b1,
                const float* __restrict__ W2, const float* __restrict__ b2,
                float* __restrict__ out) {
  __shared__ float sX[HC], sH[HC], sG[4 * HC], sT[HC / 2];
  const int tid = threadIdx.x;
  if (tid < HC) sH[tid] = 0.f;
  float cc = 0.f;
  const float bsum = bih[tid] + bhh[tid];
  const float* wi = Wih + (size_t)tid * HC;
  const float* wh = Whh + (size_t)tid * HC;
  __syncthreads();

  for (int t = 0; t < S_GRAPHS; ++t) {
    if (tid < HC) sX[tid] = pooled[(size_t)t * HC + tid];
    __syncthreads();
    float acc = bsum;
#pragma unroll 8
    for (int k = 0; k < HC; ++k) acc += wi[k] * sX[k] + wh[k] * sH[k];
    sG[tid] = acc;
    __syncthreads();
    if (tid < HC) {
      const float ig = 1.f / (1.f + __expf(-sG[tid]));
      const float fg = 1.f / (1.f + __expf(-sG[HC + tid]));
      const float gg = tanhf(sG[2 * HC + tid]);
      const float og = 1.f / (1.f + __expf(-sG[3 * HC + tid]));
      cc = fg * cc + ig * gg;
      sH[tid] = og * tanhf(cc);
    }
    __syncthreads();
  }

  if (tid < HC / 2) {
    float acc = b1[tid];
    for (int k = 0; k < HC; ++k) acc += sH[k] * W1[k * (HC / 2) + tid];
    sT[tid] = fmaxf(acc, 0.f);
  }
  __syncthreads();
  if (tid == 0) {
    float acc = b2[0];
    for (int j = 0; j < HC / 2; ++j) acc += sT[j] * W2[j];
    out[0] = acc;
  }
}

// =============== launch =============================================================
extern "C" void kernel_launch(void* const* d_in, const int* in_sizes, int n_in,
                              void* d_out, int out_size, void* d_ws, size_t ws_size,
                              hipStream_t stream) {
  const float* x     = (const float*)d_in[0];
  const int*   EI    = (const int*)d_in[1];
  const float* EA    = (const float*)d_in[2];
  const float* Wl1   = (const float*)d_in[4];
  const float* bl1   = (const float*)d_in[5];
  const float* Wr1   = (const float*)d_in[6];
  const float* br1   = (const float*)d_in[7];
  const float* We1   = (const float*)d_in[8];
  const float* att1  = (const float*)d_in[9];
  const float* bias1 = (const float*)d_in[10];
  const float* Wl2   = (const float*)d_in[11];
  const float* bl2   = (const float*)d_in[12];
  const float* Wr2   = (const float*)d_in[13];
  const float* br2   = (const float*)d_in[14];
  const float* We2   = (const float*)d_in[15];
  const float* att2  = (const float*)d_in[16];
  const float* bias2 = (const float*)d_in[17];
  const float* Wih   = (const float*)d_in[18];
  const float* Whh   = (const float*)d_in[19];
  const float* bih   = (const float*)d_in[20];
  const float* bhh   = (const float*)d_in[21];
  const float* W1    = (const float*)d_in[22];
  const float* b1    = (const float*)d_in[23];
  const float* W2    = (const float*)d_in[24];
  const float* b2    = (const float*)d_in[25];

  float* ws     = (float*)d_ws;
  float* xl1    = ws;                                    // N*192
  float* xr1    = xl1 + (size_t)N_NODES * F1;            // N*192
  float* h1     = xr1 + (size_t)N_NODES * F1;            // N*192
  float* xl2    = h1  + (size_t)N_NODES * F1;            // N*64
  float* xr2    = xl2 + (size_t)N_NODES * HC;            // N*64
  float* pooled = xr2 + (size_t)N_NODES * HC;            // S*64
  _Float16* wpack = (_Float16*)(pooled + (size_t)S_GRAPHS * HC);  // 72*32*16 f16

  stgat_pack_w<<<72, 32, 0, stream>>>(Wl1, Wr1, Wl2, Wr2, wpack);

  stgat_gemm1<<<N_NODES / 64, 128, 0, stream>>>(x, wpack, bl1, br1, xl1, xr1);

  const size_t sh1 = (size_t)(NPG * F1 * 2 + F1 * EF + F1 + NPG * EF + NPG +
                              H1 * EFF + H1 * NPG) * 4 +
                     (size_t)(EFF * 2 + NPG * 2 + NPG + 1) * 4;
  stgat_attn1<<<S_GRAPHS, 256, sh1, stream>>>(xl1, xr1, EI, EA, We1, att1, bias1, h1);

  stgat_gemm2<<<N_NODES / 64, 128, 0, stream>>>(h1, wpack, bl2, br2, xl2, xr2);

  const size_t sh2 = (size_t)(NPG * HC * 2 + HC * EF + HC + NPG * EF + NPG +
                              EFF + NPG + HC) * 4 +
                     (size_t)(EFF * 2 + NPG * 2 + NPG + 1) * 4;
  stgat_attn2<<<S_GRAPHS, 256, sh2, stream>>>(xl2, xr2, EI, EA, We2, att2, bias2, pooled);

  stgat_lstm_head<<<1, 256, 0, stream>>>(pooled, Wih, Whh, bih, bhh, W1, b1, W2, b2,
                                         (float*)d_out);
}
